// EmotionLSTM_13469017440930
// MI455X (gfx1250) — matrix operand order repaired and checked by reading, not compiled, not run on hardware
//
#include <hip/hip_runtime.h>
#include <hip/hip_bf16.h>

// Problem constants
#define BB 256
#define TT 512
#define DD 300
#define HH 128
#define GG 512   // 4*H
#define CC 6

typedef _Float16 v8h  __attribute__((ext_vector_type(8)));
typedef _Float16 v16h __attribute__((ext_vector_type(16)));
typedef float    v8f  __attribute__((ext_vector_type(8)));

// ---- WMMA helpers (CDNA5 16x16x32 f16 -> f32) -------------------------------
__device__ __forceinline__ v8f wmma_f16(v16h a, v16h b, v8f c) {
  // (neg_a, A, neg_b, B, c_mod, C, reuse_a, reuse_b)
  return __builtin_amdgcn_wmma_f32_16x16x32_f16(false, a, false, b, (short)0, c,
                                                false, false);
}

// A fragment (16x32 f16, MxK) from row-major LDS [m][k], row stride `stride` halfs.
// Lane L: m = L&15; halfs 0-7 = K k0+base+0..7, halfs 8-15 = K k0+base+16..23,
// base = (L<16 ? 0 : 8).  (ISA 7.12.2 "16-bit A-Matrix 16x32")
__device__ __forceinline__ v16h load_afrag(const _Float16* s, int m, int stride,
                                           int k0, int hi) {
  const _Float16* p = s + m * stride + k0 + hi * 8;
  v8h lo = *(const v8h*)(p);
  v8h up = *(const v8h*)(p + 16);
  return __builtin_shufflevector(lo, up, 0, 1, 2, 3, 4, 5, 6, 7, 8, 9, 10, 11,
                                 12, 13, 14, 15);
}

// B fragment (32x16 f16, KxN) where memory holds W[n][k] row-major (stride halfs).
// Lane L: n = L&15 col; halfs 0-7 = K k0+base+0..7, halfs 8-15 = K k0+base+16..23, base=(L<16?0:8).
__device__ __forceinline__ v16h load_bfrag(const _Float16* s, int n, int stride,
                                           int k0, int hi) {
  const _Float16* p = s + n * stride + k0 + hi * 8;
  v8h lo = *(const v8h*)(p);
  v8h up = *(const v8h*)(p + 16);
  return __builtin_shufflevector(lo, up, 0, 1, 2, 3, 4, 5, 6, 7, 8, 9, 10, 11,
                                 12, 13, 14, 15);
}

// ---- Fast activations (gfx1250 trans ops, co-execute with XDL WMMA) ---------
__device__ __forceinline__ float fast_exp2(float x) {
#if __has_builtin(__builtin_amdgcn_exp2f)
  return __builtin_amdgcn_exp2f(x);
#else
  return __expf(x * 0.69314718056f);
#endif
}
__device__ __forceinline__ float fast_rcp(float x) {
#if __has_builtin(__builtin_amdgcn_rcpf)
  return __builtin_amdgcn_rcpf(x);
#else
  return 1.0f / x;
#endif
}
__device__ __forceinline__ float fast_sigmoid(float x) {
  return fast_rcp(1.0f + fast_exp2(-1.44269504089f * x));
}
__device__ __forceinline__ float fast_tanh(float x) {
#if __has_builtin(__builtin_amdgcn_tanhf)
  return __builtin_amdgcn_tanhf(x);
#else
  return 2.0f * fast_sigmoid(2.0f * x) - 1.0f;
#endif
}

// gx workspace layout: WMMA C-tile order, so phase-2 seeds each 16x16 gate tile
// with ONE global_load_b128 per lane-contiguous 512B wave transaction:
//   tile_index = (t*16 + bt)*32 + ntile        (bt = b/16, ntile = n/16)
//   element    = tile_index*256 + lane*8 + j   (lane = C-layout lane, j = vgpr)
// f16 total = 512*256*512*2 = 128 MB -> resident in MI455X's 192 MB L2.

// ---- Phase 1: gx = x @ W_ih^T + b_ih + b_hh ---------------------------------
// Block: 256 threads = 8 waves, tile = 64 rows (r = t*256+b) x 512 cols,
// K = 300 zero-padded to 320 (10 WMMA k-steps).
__global__ __launch_bounds__(256, 1) void lstm_gx_kernel(
    const float* __restrict__ x, const float* __restrict__ Wih,
    const float* __restrict__ bih, const float* __restrict__ bhh,
    _Float16* __restrict__ gxs) {
  extern __shared__ char smem[];
  _Float16* sX = (_Float16*)smem;  // 64 x 320 halfs (zero-padded K)
  _Float16* sW = sX + 64 * 320;    // 512 x 32 halfs (current K-chunk)

  const int tid = threadIdx.x;
  const int r0 = blockIdx.x * 64;

  // Stage x tile (f32 -> f16, zero-pad K to 320)
  for (int idx = tid; idx < 64 * 320; idx += 256) {
    int row = idx / 320, k = idx - row * 320;
    int r = r0 + row;
    int b = r & 255, t = r >> 8;
    float v = (k < DD) ? x[((size_t)b * TT + t) * DD + k] : 0.0f;
    sX[row * 320 + k] = (_Float16)v;
  }

  const int lane = tid & 31, w = tid >> 5;
  const int l16 = lane & 15, hi = lane >> 4;
  const int mt = w & 3;   // M-tile 0..3 (16 rows each)
  const int nh = w >> 2;  // 0/1 -> cols [nh*256, nh*256+256)

  v8f acc[16];
#pragma unroll
  for (int i = 0; i < 16; ++i) {
    v8f z = {0.f, 0.f, 0.f, 0.f, 0.f, 0.f, 0.f, 0.f};
    acc[i] = z;
  }

  for (int kt = 0; kt < 10; ++kt) {
    const int k0 = kt * 32;
    __syncthreads();  // prior-iteration sW reads done (also covers sX stage)
    for (int idx = tid; idx < 512 * 32; idx += 256) {
      int n = idx >> 5, kk = idx & 31, k = k0 + kk;
      float v = (k < DD) ? Wih[n * DD + k] : 0.0f;
      sW[idx] = (_Float16)v;
    }
    __syncthreads();

    v16h a = load_afrag(sX, mt * 16 + l16, 320, k0, hi);
#pragma unroll
    for (int nt = 0; nt < 16; ++nt) {
      int n = nh * 256 + nt * 16 + l16;
      v16h bf = load_bfrag(sW, n, 32, 0, hi);
      acc[nt] = wmma_f16(a, bf, acc[nt]);
    }
  }

  // Add bias, convert to f16, store one b128 per tile in C-tile order
  const int t = r0 >> 8;
  const int bt = ((r0 & 255) >> 4) + mt;  // batch-tile index 0..15
#pragma unroll
  for (int nt = 0; nt < 16; ++nt) {
    int ntile = nh * 16 + nt;
    int n = ntile * 16 + l16;
    float bias = bih[n] + bhh[n];
    v8h hv;
#pragma unroll
    for (int j = 0; j < 8; ++j) hv[j] = (_Float16)(acc[nt][j] + bias);
    size_t tile = ((size_t)t * 16 + bt) * 32 + ntile;
    *(v8h*)(gxs + tile * 256 + lane * 8) = hv;
  }
}

// ---- Phase 2: sequential LSTM scan + FC head --------------------------------
// Grid: 16 blocks (batch tiles of 16), 256 threads = 8 waves.
// Wave w owns hidden cols [w*16, w*16+16): gate tiles at n = gate*128 + w*16.
// W_hh^T fragments hoisted to registers (128 VGPRs) after one LDS staging;
// h broadcast via double-buffered LDS tile, one s_barrier per timestep;
// gx tile for step t+1 prefetched during step t's WMMA/activation work.
__global__ __launch_bounds__(256, 1) void lstm_rec_kernel(
    const _Float16* __restrict__ gxs, const float* __restrict__ Whh,
    const int* __restrict__ lengths, const float* __restrict__ fcw,
    const float* __restrict__ fcb, float* __restrict__ out) {
  extern __shared__ char smem[];
  _Float16* sW = (_Float16*)smem;  // 512 x 128 halfs = 128KB
  _Float16* sh = sW + GG * HH;     // 2 x (16 x 128) halfs (h double buffer)

  const int tid = threadIdx.x;
  const int bt = blockIdx.x;       // batch tile (16 rows)
  const int b0 = bt * 16;

  for (int idx = tid; idx < GG * HH; idx += 256) sW[idx] = (_Float16)Whh[idx];
  for (int idx = tid; idx < 16 * HH; idx += 256) sh[idx] = (_Float16)0.0f;
  __syncthreads();

  const int lane = tid & 31, w = tid >> 5;
  const int l16 = lane & 15, hi = lane >> 4;
  const int hc0 = w * 16;

  // Resident B fragments: [gate][kstep] of W_hh^T (n = gate*128+hc0 block)
  v16h Bf[4][4];
#pragma unroll
  for (int g = 0; g < 4; ++g)
#pragma unroll
    for (int ks = 0; ks < 4; ++ks)
      Bf[g][ks] = load_bfrag(sW, g * 128 + hc0 + l16, HH, ks * 32, hi);

  int len[8];
  float c[8], h[8];
#pragma unroll
  for (int j = 0; j < 8; ++j) {
    len[j] = lengths[b0 + hi * 8 + j];
    c[j] = 0.0f;
    h[j] = 0.0f;
  }

  // Prefetch gx tiles for t = 0 (one b128 per gate)
  v8h pre[4];
#pragma unroll
  for (int g = 0; g < 4; ++g) {
    size_t tile = ((size_t)0 * 16 + bt) * 32 + (g * 8 + w);
    pre[g] = *(const v8h*)(gxs + tile * 256 + lane * 8);
  }

  for (int t = 0; t < TT; ++t) {
    const _Float16* shr = sh + (t & 1) * (16 * HH);

    // Seed accumulators from prefetched gx tiles
    v8f acc[4];
#pragma unroll
    for (int g = 0; g < 4; ++g)
#pragma unroll
      for (int j = 0; j < 8; ++j) acc[g][j] = (float)pre[g][j];

    // Prefetch gx for t+1 (latency hidden behind WMMA + activations)
    {
      int tn = (t + 1 < TT) ? (t + 1) : t;
#pragma unroll
      for (int g = 0; g < 4; ++g) {
        size_t tile = ((size_t)tn * 16 + bt) * 32 + (g * 8 + w);
        pre[g] = *(const v8h*)(gxs + tile * 256 + lane * 8);
      }
    }

    // gates += h @ W_hh^T  (K=128 in 4 WMMA k-steps, A shared across gates)
#pragma unroll
    for (int ks = 0; ks < 4; ++ks) {
      v16h a = load_afrag(shr, l16, HH, ks * 32, hi);
#pragma unroll
      for (int g = 0; g < 4; ++g) acc[g] = wmma_f16(a, Bf[g][ks], acc[g]);
    }

    _Float16* shw = sh + ((t + 1) & 1) * (16 * HH);
#pragma unroll
    for (int j = 0; j < 8; ++j) {
      float iv = fast_sigmoid(acc[0][j]);
      float fv = fast_sigmoid(acc[1][j]);
      float gv = fast_tanh(acc[2][j]);
      float ov = fast_sigmoid(acc[3][j]);
      float cn = fv * c[j] + iv * gv;
      float hn = ov * fast_tanh(cn);
      bool mk = t < len[j];
      c[j] = mk ? cn : c[j];
      h[j] = mk ? hn : h[j];
      shw[(hi * 8 + j) * HH + hc0 + l16] = (_Float16)h[j];
    }
    __syncthreads();
  }

  // FC head: out[b][cls] = h[b][:] . fc_w[cls][:] + fc_b[cls]; final h in buf 0
  if (w == 0) {
    const _Float16* hf = sh;  // buffer (TT & 1) == 0
    for (int p = lane; p < 16 * CC; p += 32) {
      int b = p / CC, cls = p - b * CC;
      float a = fcb[cls];
      for (int k = 0; k < HH; ++k)
        a += (float)hf[b * HH + k] * fcw[cls * HH + k];
      out[(size_t)(b0 + b) * CC + cls] = a;
    }
  }
}

extern "C" void kernel_launch(void* const* d_in, const int* in_sizes, int n_in,
                              void* d_out, int out_size, void* d_ws,
                              size_t ws_size, hipStream_t stream) {
  const float* x   = (const float*)d_in[0];
  const float* Wih = (const float*)d_in[1];
  const float* Whh = (const float*)d_in[2];
  const float* bih = (const float*)d_in[3];
  const float* bhh = (const float*)d_in[4];
  const float* fcw = (const float*)d_in[5];
  const float* fcb = (const float*)d_in[6];
  const int* lens  = (const int*)d_in[7];
  float* outp      = (float*)d_out;

  _Float16* gxs = (_Float16*)d_ws;  // 128 MB, WMMA C-tile layout

  size_t shm1 = (size_t)(64 * 320 + 512 * 32) * sizeof(_Float16);    // 72 KB
  size_t shm2 = (size_t)(GG * HH + 2 * 16 * HH) * sizeof(_Float16);  // 136 KB
  hipFuncSetAttribute(reinterpret_cast<const void*>(lstm_gx_kernel),
                      hipFuncAttributeMaxDynamicSharedMemorySize, (int)shm1);
  hipFuncSetAttribute(reinterpret_cast<const void*>(lstm_rec_kernel),
                      hipFuncAttributeMaxDynamicSharedMemorySize, (int)shm2);

  lstm_gx_kernel<<<(TT * BB) / 64, 256, shm1, stream>>>(x, Wih, bih, bhh, gxs);
  lstm_rec_kernel<<<BB / 16, 256, shm2, stream>>>(gxs, Whh, lens, fcw, fcb,
                                                  outp);
}
